// S2VTWithAttention_43336220016641
// MI455X (gfx1250) — matrix operand order repaired and checked by reading, not compiled, not run on hardware
//
#include <hip/hip_runtime.h>

// ---------------------------------------------------------------------------
// CDNA5 (gfx1250) S2VT-with-attention forward.
// Heavy GEMMs use v_wmma_f32_16x16x32_bf16 with 32x64 wave tiles and
// software-pipelined fragment loads (grouped B loads + next-k A prefetch).
// ---------------------------------------------------------------------------

typedef __attribute__((ext_vector_type(16))) __bf16 v16bf;
typedef __attribute__((ext_vector_type(8)))  float  v8f;

union Frag16 { v16bf v; uint4 u[2]; };

// ---- constants -------------------------------------------------------------
#define BB 64
#define TT 80
#define FF 4096
#define HH 512
#define VV 32000
#define LL 31
#define G4 (4 * HH)      // 2048

__device__ __forceinline__ unsigned short f2bf(float f) {
  unsigned u = __float_as_uint(f);
  u += 0x7FFFu + ((u >> 16) & 1u);   // round-to-nearest-even
  return (unsigned short)(u >> 16);
}

__device__ __forceinline__ float sigmoidf_(float x) {
  return 1.0f / (1.0f + __expf(-x));
}

__device__ __forceinline__ v8f wmma_bf16(v16bf a, v16bf b, v8f c) {
  return __builtin_amdgcn_wmma_f32_16x16x32_bf16(false, a, false, b, (short)0, c,
                                                 false, false);
}

// A-fragment: 16x32 bf16, A row-major [M,K] (lda elems).
// lane<16 -> M=lane, K = {k0..k0+7, k0+16..k0+23}; lane>=16 -> +8 K offset.
__device__ __forceinline__ v16bf load_afrag(const unsigned short* A, long lda,
                                            int row, int k0, int lane) {
  Frag16 a;
  const unsigned short* p = A + (long)row * lda + k0 + ((lane >> 4) << 3);
  a.u[0] = *(const uint4*)(p);
  a.u[1] = *(const uint4*)(p + 16);
  return a.v;
}

// B-fragment: 32x16 bf16 where B[k][n] = W[n][k], W row-major [N,K] (ldw elems).
// lane holds column n (= W row n), K = {k0..k0+7, k0+16..k0+23}; lane>=16 -> +8 K offset.
__device__ __forceinline__ v16bf load_bfrag(const unsigned short* W, long ldw,
                                            int nrow, int k0, int lane) {
  Frag16 b;
  const unsigned short* p = W + (long)nrow * ldw + k0 + ((lane >> 4) << 3);
  b.u[0] = *(const uint4*)(p);
  b.u[1] = *(const uint4*)(p + 16);
  return b.v;
}

// ---------------------------------------------------------------------------
// Generic GEMM: C[M,N] = A[M,K] @ W[N,K]^T (+bias[N]).  f32 out.
// Block = 128 threads (4 waves). Wave tile = 32(M) x 64(N); block tile 32x256.
// Software pipelined: grouped B loads, A prefetch one k-step ahead.
// Requires: M%32==0, N%256==0, K%32==0.  grid = (N/256, M/32).
// ---------------------------------------------------------------------------
__global__ void gemm_bf16_wmma(const unsigned short* __restrict__ A, long lda,
                               const unsigned short* __restrict__ W, long ldw,
                               float* __restrict__ C, long ldc,
                               const float* __restrict__ bias, int K) {
  const int lane = threadIdx.x & 31;
  const int wave = threadIdx.x >> 5;
  const int m0 = blockIdx.y << 5;
  const int n0 = (blockIdx.x << 8) + (wave << 6);
  const int ar0 = m0 + (lane & 15);
  const int ar1 = ar0 + 16;
  const int bcol = lane & 15;

  v8f acc[8] = {};
  v16bf a0 = load_afrag(A, lda, ar0, 0, lane);
  v16bf a1 = load_afrag(A, lda, ar1, 0, lane);

  for (int k0 = 0; k0 < K; k0 += 32) {
    // group all B loads for this k-step
    v16bf b0 = load_bfrag(W, ldw, n0 +  0 + bcol, k0, lane);
    v16bf b1 = load_bfrag(W, ldw, n0 + 16 + bcol, k0, lane);
    v16bf b2 = load_bfrag(W, ldw, n0 + 32 + bcol, k0, lane);
    v16bf b3 = load_bfrag(W, ldw, n0 + 48 + bcol, k0, lane);
    // prefetch next k-step's A while WMMAs run
    const int kn = (k0 + 32 < K) ? (k0 + 32) : k0;
    v16bf a0n = load_afrag(A, lda, ar0, kn, lane);
    v16bf a1n = load_afrag(A, lda, ar1, kn, lane);

    acc[0] = wmma_bf16(a0, b0, acc[0]);
    acc[4] = wmma_bf16(a1, b0, acc[4]);
    acc[1] = wmma_bf16(a0, b1, acc[1]);
    acc[5] = wmma_bf16(a1, b1, acc[5]);
    acc[2] = wmma_bf16(a0, b2, acc[2]);
    acc[6] = wmma_bf16(a1, b2, acc[6]);
    acc[3] = wmma_bf16(a0, b3, acc[3]);
    acc[7] = wmma_bf16(a1, b3, acc[7]);

    a0 = a0n;
    a1 = a1n;
  }

#pragma unroll
  for (int h = 0; h < 2; ++h) {
    const int rb = m0 + (h << 4) + ((lane >> 4) << 3);
#pragma unroll
    for (int jt = 0; jt < 4; ++jt) {
      const int col = n0 + jt * 16 + (lane & 15);
      const float bv = bias ? bias[col] : 0.0f;
#pragma unroll
      for (int r = 0; r < 8; ++r) {
        C[(long)(rb + r) * ldc + col] = acc[h * 4 + jt][r] + bv;
      }
    }
  }
}

// ---------------------------------------------------------------------------
// Encoder LSTM step (fused gates GEMM + activations).
// gates = Xproj_t (precomputed x@Wih^T + bih + bhh) + h@Whh^T.
// Block = 1 wave; block owns 16(batch) x 16(hidden) outputs across all 4 gates.
// grid = (H/16 = 32, B/16 = 4).
// ---------------------------------------------------------------------------
__global__ void enc_step_kernel(const unsigned short* __restrict__ h_in,   // [64,512] bf16
                                const unsigned short* __restrict__ Whh,    // [2048,512] bf16
                                const float* __restrict__ Xp, long xp_stride,
                                float* __restrict__ c,                     // [64,512] f32
                                unsigned short* __restrict__ h_out,        // [64,512] bf16
                                float* __restrict__ eo,                    // enc_out f32 slice (t)
                                unsigned short* __restrict__ eob,          // enc_out bf16 slice (t)
                                long eo_stride) {
  const int lane = threadIdx.x;
  const int b0 = blockIdx.y << 4;
  const int j0 = blockIdx.x << 4;
  const int arow = b0 + (lane & 15);
  const int bcol = lane & 15;

  v8f acc[4] = {};
  v16bf a = load_afrag(h_in, HH, arow, 0, lane);
  for (int k0 = 0; k0 < HH; k0 += 32) {
    v16bf bf[4];
#pragma unroll
    for (int g = 0; g < 4; ++g)
      bf[g] = load_bfrag(Whh, HH, g * HH + j0 + bcol, k0, lane);
    const int kn = (k0 + 32 < HH) ? (k0 + 32) : k0;
    v16bf an = load_afrag(h_in, HH, arow, kn, lane);
#pragma unroll
    for (int g = 0; g < 4; ++g) acc[g] = wmma_bf16(a, bf[g], acc[g]);
    a = an;
  }

  const int rb = b0 + ((lane >> 4) << 3);
  const int j = j0 + (lane & 15);
#pragma unroll
  for (int r = 0; r < 8; ++r) {
    const long b = rb + r;
    const float* xr = Xp + b * xp_stride;
    float gi = acc[0][r] + xr[j];
    float gf = acc[1][r] + xr[HH + j];
    float gg = acc[2][r] + xr[2 * HH + j];
    float go = acc[3][r] + xr[3 * HH + j];
    float cn = sigmoidf_(gf) * c[b * HH + j] + sigmoidf_(gi) * tanhf(gg);
    float hn = sigmoidf_(go) * tanhf(cn);
    c[b * HH + j] = cn;
    h_out[b * HH + j] = f2bf(hn);
    eo[b * eo_stride + j] = hn;
    eob[b * eo_stride + j] = f2bf(hn);
  }
}

// ---------------------------------------------------------------------------
// Decoder LSTM step: gates = xcat@Wih^T + h@Whh^T + bias.  Same tiling.
// ---------------------------------------------------------------------------
__global__ void dec_step_kernel(const unsigned short* __restrict__ h_in,   // [64,512] bf16
                                const unsigned short* __restrict__ xcat,   // [64,1024] bf16
                                const unsigned short* __restrict__ Wih,    // [2048,1024] bf16
                                const unsigned short* __restrict__ Whh,    // [2048,512] bf16
                                const float* __restrict__ bias,            // [2048] f32
                                float* __restrict__ c,                     // [64,512] f32
                                unsigned short* __restrict__ h_out) {      // [64,512] bf16
  const int lane = threadIdx.x;
  const int b0 = blockIdx.y << 4;
  const int j0 = blockIdx.x << 4;
  const int arow = b0 + (lane & 15);
  const int bcol = lane & 15;

  v8f acc[4] = {};
  {
    v16bf a = load_afrag(xcat, 2 * HH, arow, 0, lane);
    for (int k0 = 0; k0 < 2 * HH; k0 += 32) {      // x part, K = 1024
      v16bf bf[4];
#pragma unroll
      for (int g = 0; g < 4; ++g)
        bf[g] = load_bfrag(Wih, 2 * HH, g * HH + j0 + bcol, k0, lane);
      const int kn = (k0 + 32 < 2 * HH) ? (k0 + 32) : k0;
      v16bf an = load_afrag(xcat, 2 * HH, arow, kn, lane);
#pragma unroll
      for (int g = 0; g < 4; ++g) acc[g] = wmma_bf16(a, bf[g], acc[g]);
      a = an;
    }
  }
  {
    v16bf a = load_afrag(h_in, HH, arow, 0, lane);
    for (int k0 = 0; k0 < HH; k0 += 32) {          // h part, K = 512
      v16bf bf[4];
#pragma unroll
      for (int g = 0; g < 4; ++g)
        bf[g] = load_bfrag(Whh, HH, g * HH + j0 + bcol, k0, lane);
      const int kn = (k0 + 32 < HH) ? (k0 + 32) : k0;
      v16bf an = load_afrag(h_in, HH, arow, kn, lane);
#pragma unroll
      for (int g = 0; g < 4; ++g) acc[g] = wmma_bf16(a, bf[g], acc[g]);
      a = an;
    }
  }

  const int rb = b0 + ((lane >> 4) << 3);
  const int j = j0 + (lane & 15);
#pragma unroll
  for (int r = 0; r < 8; ++r) {
    const long b = rb + r;
    float gi = acc[0][r] + bias[j];
    float gf = acc[1][r] + bias[HH + j];
    float gg = acc[2][r] + bias[2 * HH + j];
    float go = acc[3][r] + bias[3 * HH + j];
    float cn = sigmoidf_(gf) * c[b * HH + j] + sigmoidf_(gi) * tanhf(gg);
    float hn = sigmoidf_(go) * tanhf(cn);
    c[b * HH + j] = cn;
    h_out[b * HH + j] = f2bf(hn);
  }
}

// ---------------------------------------------------------------------------
// scores[b,t] = sum_j v_w[j] * tanh(hproj[b,j] + enc_proj[b*T+t, j])
// grid = B*T blocks, 128 threads.
// ---------------------------------------------------------------------------
__global__ void attn_scores_kernel(const float* __restrict__ hproj,
                                   const float* __restrict__ enc_proj,
                                   const float* __restrict__ v_w,
                                   float* __restrict__ scores) {
  __shared__ float red[128];
  const int bt = blockIdx.x;
  const int b = bt / TT;
  float s = 0.0f;
  for (int j = threadIdx.x; j < HH; j += 128)
    s += v_w[j] * tanhf(hproj[(long)b * HH + j] + enc_proj[(long)bt * HH + j]);
  red[threadIdx.x] = s;
  __syncthreads();
  for (int off = 64; off > 0; off >>= 1) {
    if ((int)threadIdx.x < off) red[threadIdx.x] += red[threadIdx.x + off];
    __syncthreads();
  }
  if (threadIdx.x == 0) scores[bt] = red[0];
}

// ---------------------------------------------------------------------------
// Per-batch softmax over T, context vector, and embedding gather.
// xcat[b] = [ emb[tok] (bf16, 512) | ctx (bf16, 512) ].  grid = B, 256 thr.
// ---------------------------------------------------------------------------
__global__ void softmax_ctx_kernel(const float* __restrict__ scores,
                                   const float* __restrict__ enc_out, // [B,T,H] f32
                                   const unsigned short* __restrict__ emb_bf,
                                   const int* __restrict__ captions,  // [B,L]
                                   int step,
                                   unsigned short* __restrict__ xcat) {
  __shared__ float w[TT];
  __shared__ float invsum;
  const int b = blockIdx.x;
  const int tid = threadIdx.x;

  if (tid < TT) w[tid] = scores[b * TT + tid];
  __syncthreads();
  if (tid == 0) {
    float m = -3.4e38f;
    for (int t = 0; t < TT; ++t) m = fmaxf(m, w[t]);
    float s = 0.0f;
    for (int t = 0; t < TT; ++t) { w[t] = __expf(w[t] - m); s += w[t]; }
    invsum = 1.0f / s;
  }
  __syncthreads();

  const float inv = invsum;
  for (int j = tid; j < HH; j += 256) {
    float acc = 0.0f;
    const float* eb = enc_out + (long)b * TT * HH + j;
    for (int t = 0; t < TT; ++t) acc += w[t] * eb[(long)t * HH];
    xcat[(long)b * (2 * HH) + HH + j] = f2bf(acc * inv);
  }
  const int tok = captions[b * LL + step];
  for (int j = tid; j < HH; j += 256)
    xcat[(long)b * (2 * HH) + j] = emb_bf[(long)tok * HH + j];
}

// ---- small utility kernels -------------------------------------------------
__global__ void cast_bf16_kernel(const float* __restrict__ s,
                                 unsigned short* __restrict__ d, long n) {
  long i = (long)blockIdx.x * blockDim.x + threadIdx.x;
  const long stride = (long)gridDim.x * blockDim.x;
  for (; i < n; i += stride) d[i] = f2bf(s[i]);
}

// Cast a column slice of a row-major matrix: d[r,c] = bf16(s[r*sstride+col0+c])
__global__ void cast_split_kernel(const float* __restrict__ s, long sstride, long col0,
                                  unsigned short* __restrict__ d, long rows, long cols) {
  long i = (long)blockIdx.x * blockDim.x + threadIdx.x;
  const long stride = (long)gridDim.x * blockDim.x;
  const long n = rows * cols;
  for (; i < n; i += stride) {
    const long r = i / cols, c = i - r * cols;
    d[i] = f2bf(s[r * sstride + col0 + c]);
  }
}

__global__ void add_bias_kernel(const float* __restrict__ a,
                                const float* __restrict__ b,
                                float* __restrict__ d, int n) {
  int i = blockIdx.x * blockDim.x + threadIdx.x;
  if (i < n) d[i] = a[i] + b[i];
}

// ---------------------------------------------------------------------------
// Host orchestration.
// ---------------------------------------------------------------------------
extern "C" void kernel_launch(void* const* d_in, const int* in_sizes, int n_in,
                              void* d_out, int out_size, void* d_ws, size_t ws_size,
                              hipStream_t stream) {
  (void)in_sizes; (void)n_in; (void)out_size; (void)ws_size;

  const float* vidf     = (const float*)d_in[0];
  const int*   captions = (const int*)d_in[1];
  const float* emb      = (const float*)d_in[2];
  const float* enc_Wih  = (const float*)d_in[3];
  const float* enc_Whh  = (const float*)d_in[4];
  const float* enc_bih  = (const float*)d_in[5];
  const float* enc_bhh  = (const float*)d_in[6];
  const float* attn_W   = (const float*)d_in[7];
  const float* attn_b   = (const float*)d_in[8];
  const float* v_w      = (const float*)d_in[9];
  const float* dec_Wih  = (const float*)d_in[10];
  const float* dec_Whh  = (const float*)d_in[11];
  const float* dec_bih  = (const float*)d_in[12];
  const float* dec_bhh  = (const float*)d_in[13];
  const float* fc_W     = (const float*)d_in[14];
  const float* fc_b     = (const float*)d_in[15];
  float* out = (float*)d_out;

  // ---- carve workspace ----
  char* wp = (char*)d_ws;
  auto carve = [&](size_t bytes) {
    char* p = wp;
    wp += (bytes + 255) & ~(size_t)255;
    return p;
  };
  unsigned short* vid_bf    = (unsigned short*)carve((size_t)BB * TT * FF * 2);
  unsigned short* encWih_bf = (unsigned short*)carve((size_t)G4 * FF * 2);
  unsigned short* encWhh_bf = (unsigned short*)carve((size_t)G4 * HH * 2);
  unsigned short* decWih_bf = (unsigned short*)carve((size_t)G4 * 2 * HH * 2);
  unsigned short* decWhh_bf = (unsigned short*)carve((size_t)G4 * HH * 2);
  unsigned short* W1_bf     = (unsigned short*)carve((size_t)HH * HH * 2);
  unsigned short* W2_bf     = (unsigned short*)carve((size_t)HH * HH * 2);
  unsigned short* fcW_bf    = (unsigned short*)carve((size_t)VV * HH * 2);
  unsigned short* emb_bf    = (unsigned short*)carve((size_t)VV * HH * 2);
  float*          Xproj     = (float*)carve((size_t)BB * TT * G4 * 4);
  float*          enc_out   = (float*)carve((size_t)BB * TT * HH * 4);
  unsigned short* encout_bf = (unsigned short*)carve((size_t)BB * TT * HH * 2);
  float*          enc_proj  = (float*)carve((size_t)BB * TT * HH * 4);
  unsigned short* hbf0      = (unsigned short*)carve((size_t)BB * HH * 2);
  unsigned short* hbf1      = (unsigned short*)carve((size_t)BB * HH * 2);
  float*          c_buf     = (float*)carve((size_t)BB * HH * 4);
  float*          hproj     = (float*)carve((size_t)BB * HH * 4);
  float*          scores    = (float*)carve((size_t)BB * TT * 4);
  unsigned short* xcat      = (unsigned short*)carve((size_t)BB * 2 * HH * 2);
  float*          bias_enc  = (float*)carve((size_t)G4 * 4);
  float*          bias_dec  = (float*)carve((size_t)G4 * 4);
  unsigned short* hbf[2] = {hbf0, hbf1};

  // ---- one-time casts / bias sums ----
  cast_bf16_kernel<<<4096, 256, 0, stream>>>(vidf, vid_bf, (long)BB * TT * FF);
  cast_bf16_kernel<<<4096, 256, 0, stream>>>(enc_Wih, encWih_bf, (long)G4 * FF);
  cast_bf16_kernel<<<1024, 256, 0, stream>>>(enc_Whh, encWhh_bf, (long)G4 * HH);
  cast_bf16_kernel<<<2048, 256, 0, stream>>>(dec_Wih, decWih_bf, (long)G4 * 2 * HH);
  cast_bf16_kernel<<<1024, 256, 0, stream>>>(dec_Whh, decWhh_bf, (long)G4 * HH);
  cast_bf16_kernel<<<4096, 256, 0, stream>>>(fc_W, fcW_bf, (long)VV * HH);
  cast_bf16_kernel<<<4096, 256, 0, stream>>>(emb, emb_bf, (long)VV * HH);
  cast_split_kernel<<<512, 256, 0, stream>>>(attn_W, 2 * HH, 0,  W1_bf, HH, HH);
  cast_split_kernel<<<512, 256, 0, stream>>>(attn_W, 2 * HH, HH, W2_bf, HH, HH);
  add_bias_kernel<<<(G4 + 255) / 256, 256, 0, stream>>>(enc_bih, enc_bhh, bias_enc, G4);
  add_bias_kernel<<<(G4 + 255) / 256, 256, 0, stream>>>(dec_bih, dec_bhh, bias_dec, G4);

  // ---- hoisted encoder input projection: Xproj = vid @ enc_Wih^T + (bih+bhh) ----
  // [5120, 2048] = [5120, 4096] @ [2048, 4096]^T
  gemm_bf16_wmma<<<dim3(G4 / 256, (BB * TT) / 32), 128, 0, stream>>>(
      vid_bf, FF, encWih_bf, FF, Xproj, G4, bias_enc, FF);

  // ---- init recurrent state ----
  hipMemsetAsync(hbf0, 0, (size_t)BB * HH * 2, stream);
  hipMemsetAsync(c_buf, 0, (size_t)BB * HH * 4, stream);

  // ---- encoder recurrence (80 steps, h ping-pong in bf16) ----
  for (int t = 0; t < TT; ++t) {
    enc_step_kernel<<<dim3(HH / 16, BB / 16), 32, 0, stream>>>(
        hbf[t & 1], encWhh_bf, Xproj + (long)t * G4, (long)TT * G4,
        c_buf, hbf[(t + 1) & 1],
        enc_out + (long)t * HH, encout_bf + (long)t * HH, (long)TT * HH);
  }

  // ---- step-invariant attention half: enc_proj = enc_out @ W2^T + attn_b ----
  gemm_bf16_wmma<<<dim3(HH / 256, (BB * TT) / 32), 128, 0, stream>>>(
      encout_bf, HH, W2_bf, HH, enc_proj, HH, attn_b, HH);

  // ---- decoder: 30 steps ----
  for (int s = 0; s < LL - 1; ++s) {
    const unsigned short* hcur = hbf[s & 1];       // encoder ended writing hbf[0]
    unsigned short* hnext = hbf[(s + 1) & 1];

    // hproj = h @ W1^T
    gemm_bf16_wmma<<<dim3(HH / 256, BB / 32), 128, 0, stream>>>(
        hcur, HH, W1_bf, HH, hproj, HH, nullptr, HH);
    // scores[b,t]
    attn_scores_kernel<<<BB * TT, 128, 0, stream>>>(hproj, enc_proj, v_w, scores);
    // softmax over t, ctx, embedding gather -> xcat
    softmax_ctx_kernel<<<BB, 256, 0, stream>>>(scores, enc_out, emb_bf, captions, s, xcat);
    // decoder LSTM step
    dec_step_kernel<<<dim3(HH / 16, BB / 16), 32, 0, stream>>>(
        hcur, xcat, decWih_bf, decWhh_bf, bias_dec, c_buf, hnext);
    // logits -> d_out[b, s, :]  (row stride = (L-1)*V), fc_W stays hot in L2
    gemm_bf16_wmma<<<dim3(VV / 256, BB / 32), 128, 0, stream>>>(
        hnext, HH, fcW_bf, HH, out + (long)s * VV, (long)(LL - 1) * VV, fc_b, HH);
  }
}
